// Attention_12197707121249
// MI455X (gfx1250) — matrix operand order repaired and checked by reading, not compiled, not run on hardware
//
#include <hip/hip_runtime.h>
#include <hip/hip_bf16.h>
#include <math.h>

// ---------------------------------------------------------------------------
// MI455X (gfx1250, wave32) implementation.
// All heavy GEMMs (QKV, QK^T, PV, proj) run through v_wmma_f32_16x16x32_f16
// with f32 accumulation. Workspace requirement: ~127 MB.
// GEMM wave tile enlarged to 32x64 (8 WMMAs : 12 ds_load_b128 per K-step).
// ---------------------------------------------------------------------------

typedef _Float16 f16;
typedef _Float16 h8  __attribute__((ext_vector_type(8)));
typedef _Float16 v16h __attribute__((ext_vector_type(16)));
typedef float    v8f  __attribute__((ext_vector_type(8)));

union V16 { v16h v; h8 h[2]; };

#define DIMC   1024
#define NHEAD  16
#define HDIM   64
#define NSEQ   1024
#define BATCH  8
#define TOK    (BATCH * NSEQ)      // 8192
#define TTBL   3969                 // (2*32-1)^2
#define EPS32  1.1920929e-7f

// ------------------------------ tiny kernels -------------------------------

__global__ __launch_bounds__(256) void k_f32_to_f16(const float* __restrict__ in,
                                                    f16* __restrict__ out, int n) {
  int i = blockIdx.x * 256 + threadIdx.x;
  if (i < n) out[i] = (f16)in[i];
}

// CPB MLP: tbl[t][h] = relu(coords @ w1^T + b1) @ w2^T + b2. One wave per row.
__global__ __launch_bounds__(256) void k_cpb(const float* __restrict__ rct,
                                             const float* __restrict__ w1,
                                             const float* __restrict__ b1,
                                             const float* __restrict__ w2,
                                             const float* __restrict__ b2,
                                             float* __restrict__ tbl) {
  int wave = (blockIdx.x * 256 + threadIdx.x) >> 5;
  int lane = threadIdx.x & 31;
  if (wave >= TTBL) return;
  float c0 = rct[wave * 2 + 0];
  float c1 = rct[wave * 2 + 1];
  float acc[NHEAD];
#pragma unroll
  for (int o = 0; o < NHEAD; ++o) acc[o] = 0.f;
#pragma unroll
  for (int jj = 0; jj < 16; ++jj) {
    int j = lane + jj * 32;
    float hv = fmaxf(c0 * w1[j * 2 + 0] + c1 * w1[j * 2 + 1] + b1[j], 0.f);
#pragma unroll
    for (int o = 0; o < NHEAD; ++o) acc[o] += hv * w2[o * 512 + j];
  }
#pragma unroll
  for (int o = 0; o < NHEAD; ++o)
#pragma unroll
    for (int d = 16; d >= 1; d >>= 1) acc[o] += __shfl_xor(acc[o], d);
  if (lane == 0) {
#pragma unroll
    for (int o = 0; o < NHEAD; ++o) tbl[wave * NHEAD + o] = acc[o] + b2[o];
  }
}

// Expand CPB table through relative_pos_index into f16 bias[NH][N][N].
__global__ __launch_bounds__(256) void k_bias_expand(const float* __restrict__ tbl,
                                                     const int* __restrict__ rpi,
                                                     f16* __restrict__ bias16) {
  int nm = blockIdx.x * 256 + threadIdx.x;
  if (nm >= NSEQ * NSEQ) return;
  int t = rpi[nm];
#pragma unroll
  for (int h = 0; h < NHEAD; ++h)
    bias16[(size_t)h * NSEQ * NSEQ + nm] = (f16)tbl[t * NHEAD + h];
}

// Per-token gating: routed top-2 + shared + w0 softmaxes -> masked[tok][16].
__global__ __launch_bounds__(256) void k_gates(const float* __restrict__ x,
                                               const float* __restrict__ wg,
                                               const float* __restrict__ wg0,
                                               const float* __restrict__ wg1,
                                               float* __restrict__ masked) {
  int wave = (blockIdx.x * 256 + threadIdx.x) >> 5;
  int lane = threadIdx.x & 31;
  if (wave >= TOK) return;
  const float* xr = &x[(size_t)wave * DIMC];
  float xv[32];
#pragma unroll
  for (int j = 0; j < 32; ++j) xv[j] = xr[lane + 32 * j];

  auto dotw = [&](const float* wrow) {
    float s = 0.f;
#pragma unroll
    for (int j = 0; j < 32; ++j) s += xv[j] * wrow[lane + 32 * j];
#pragma unroll
    for (int d = 16; d >= 1; d >>= 1) s += __shfl_xor(s, d);
    return s;
  };

  float g[8], s8[8], w0a, w0b;
#pragma unroll
  for (int e = 0; e < 8; ++e) g[e] = dotw(&wg[e * DIMC]);
#pragma unroll
  for (int e = 0; e < 8; ++e) s8[e] = dotw(&wg1[e * DIMC]);
  w0a = dotw(&wg0[0]);
  w0b = dotw(&wg0[DIMC]);

  // softmax over routed logits
  float mx = g[0];
#pragma unroll
  for (int e = 1; e < 8; ++e) mx = fmaxf(mx, g[e]);
  float sum = 0.f;
#pragma unroll
  for (int e = 0; e < 8; ++e) { g[e] = __expf(g[e] - mx); sum += g[e]; }
#pragma unroll
  for (int e = 0; e < 8; ++e) g[e] /= sum;
  // top-2
  int i1 = 0;
#pragma unroll
  for (int e = 1; e < 8; ++e) if (g[e] > g[i1]) i1 = e;
  int i2 = (i1 == 0) ? 1 : 0;
#pragma unroll
  for (int e = 0; e < 8; ++e) if (e != i1 && g[e] > g[i2]) i2 = e;
  float s2 = fmaxf(g[i1] + g[i2], EPS32);
  float rt[8];
#pragma unroll
  for (int e = 0; e < 8; ++e) rt[e] = (e == i1 || e == i2) ? g[e] / s2 : 0.f;

  // softmax over shared logits
  mx = s8[0];
#pragma unroll
  for (int e = 1; e < 8; ++e) mx = fmaxf(mx, s8[e]);
  sum = 0.f;
#pragma unroll
  for (int e = 0; e < 8; ++e) { s8[e] = __expf(s8[e] - mx); sum += s8[e]; }
#pragma unroll
  for (int e = 0; e < 8; ++e) s8[e] /= sum;

  // softmax over w0 pair
  float wm = fmaxf(w0a, w0b);
  float ea = __expf(w0a - wm), eb = __expf(w0b - wm);
  float w0n0 = ea / (ea + eb), w0n1 = eb / (ea + eb);

  if (lane == 0) {
    float* mrow = &masked[(size_t)wave * NHEAD];
#pragma unroll
    for (int e = 0; e < 8; ++e) mrow[e] = s8[e] * 8.f * (w0n0 * 2.f);     // shared heads
#pragma unroll
    for (int e = 0; e < 8; ++e) mrow[8 + e] = rt[e] * 2.f * (w0n1 * 2.f); // routed heads
  }
}

// ------------------------------ WMMA GEMM ----------------------------------
// C[M,Nout] = A[M,K] @ W[Nout,K]^T (+bias). 256 threads, 8 waves.
// Workgroup tile 256x64, wave tile 32x64 (2x4 accumulators), K-step 32.
// Per K-step per wave: 8 WMMAs vs 12 ds_load_b128 (B fragments reused 2x).
__global__ __launch_bounds__(256) void k_gemm_f16(const f16* __restrict__ A,
                                                  const f16* __restrict__ W,
                                                  const float* __restrict__ bias,
                                                  f16* __restrict__ O16,
                                                  float* __restrict__ O32,
                                                  int M, int Nout, int K, int store16) {
  __shared__ f16 sA[256][32];   // 16 KB
  __shared__ f16 sB[64][32];    //  4 KB
  const int tid = threadIdx.x;
  const int w = tid >> 5, lane = tid & 31;
  const int hv = lane >> 4, l16 = lane & 15;
  const int m0 = blockIdx.x * 256, n0 = blockIdx.y * 64;

  const v8f vz = {0.f, 0.f, 0.f, 0.f, 0.f, 0.f, 0.f, 0.f};
  v8f acc[2][4];
#pragma unroll
  for (int hm = 0; hm < 2; ++hm)
#pragma unroll
    for (int t = 0; t < 4; ++t) acc[hm][t] = vz;

  for (int k0 = 0; k0 < K; k0 += 32) {
    {
      // A: one thread stages one full 32-half row (64 contiguous bytes)
      const f16* src = &A[(size_t)(m0 + tid) * K + k0];
      *(h8*)&sA[tid][0]  = *(const h8*)(src + 0);
      *(h8*)&sA[tid][8]  = *(const h8*)(src + 8);
      *(h8*)&sA[tid][16] = *(const h8*)(src + 16);
      *(h8*)&sA[tid][24] = *(const h8*)(src + 24);
      if (k0 + 32 < K) __builtin_prefetch(src + 32, 0, 1);
      if (tid < 128) {
        int n = tid >> 1, kb = (tid & 1) * 16;
        const f16* sw = &W[(size_t)(n0 + n) * K + k0 + kb];
        *(h8*)&sB[n][kb]     = *(const h8*)sw;
        *(h8*)&sB[n][kb + 8] = *(const h8*)(sw + 8);
      }
    }
    __syncthreads();
    // A fragments: lane row m = 32*w + 16*hm + l16,
    // contiguous K run [16*hv, +16)
    V16 a0, a1;
    a0.h[0] = *(const h8*)&sA[32 * w + l16][16 * hv];
    a0.h[1] = *(const h8*)&sA[32 * w + l16][16 * hv + 8];
    a1.h[0] = *(const h8*)&sA[32 * w + 16 + l16][16 * hv];
    a1.h[1] = *(const h8*)&sA[32 * w + 16 + l16][16 * hv + 8];
#pragma unroll
    for (int t = 0; t < 4; ++t) {
      // B fragment: lane column n = t*16 + l16, contiguous K run [16*hv, +16)
      v16h b = *(const v16h*)&sB[t * 16 + l16][16 * hv];
      acc[0][t] = __builtin_amdgcn_wmma_f32_16x16x32_f16(false, a0.v, false, b,
                                                         (short)0, acc[0][t], false, false);
      acc[1][t] = __builtin_amdgcn_wmma_f32_16x16x32_f16(false, a1.v, false, b,
                                                         (short)0, acc[1][t], false, false);
    }
    __syncthreads();
  }

#pragma unroll
  for (int hm = 0; hm < 2; ++hm)
#pragma unroll
    for (int t = 0; t < 4; ++t) {
      int col = n0 + t * 16 + l16;
      float bv = bias ? bias[col] : 0.f;
#pragma unroll
      for (int r = 0; r < 8; ++r) {
        int row = m0 + 32 * w + 16 * hm + r + 8 * hv;
        float val = acc[hm][t][r] + bv;
        if (store16) O16[(size_t)row * Nout + col] = (f16)val;
        else         O32[(size_t)row * Nout + col] = val;
      }
    }
}

// ---------------------------- flash attention ------------------------------
// grid (N/128, NH, B), 256 threads (8 waves), each wave owns 16 query rows.
__global__ __launch_bounds__(256) void k_attn(const f16* __restrict__ qkv,
                                              const f16* __restrict__ bias16,
                                              const float* __restrict__ qe,
                                              const float* __restrict__ temp,
                                              const float* __restrict__ masked,
                                              const int* __restrict__ Hp,
                                              const int* __restrict__ Wp,
                                              f16* __restrict__ attn_out) {
  __shared__ f16 sQ[128][64];      // normalized+embedded+scaled q (f16)
  __shared__ f16 sK[64][64];       // normalized k block
  __shared__ f16 sVT[64][64];      // v block transposed [dim][col]
  __shared__ f16 sP[8][16][64];    // per-wave P tile (D-layout -> A-layout)

  const int tid = threadIdx.x;
  const int w = tid >> 5, lane = tid & 31;
  const int hv = lane >> 4, l16 = lane & 15;
  const int h = blockIdx.y, b = blockIdx.z;
  const int qrow0 = blockIdx.x * 128;
  const size_t tok0 = (size_t)b * NSEQ;

  const float seq_scale = logf((float)(Hp[0] * Wp[0]));
  const float scale = log1pf(__expf(temp[h])) * seq_scale;   // softplus * log(HW)

  // ---- stage qn = (normalize(q) + qe) * scale, 2 threads per row ----
  {
    int row = tid >> 1, c0 = (tid & 1) * 32;
    const f16* src = &qkv[(tok0 + qrow0 + row) * (3 * DIMC) + h * HDIM + c0];
    float tv[32]; float ss = 0.f;
#pragma unroll
    for (int j = 0; j < 32; ++j) { float v = (float)src[j]; tv[j] = v; ss += v * v; }
    ss += __shfl_xor(ss, 1);
    float inv = 1.f / fmaxf(sqrtf(ss), 1e-12f);
    const float* qeh = &qe[h * HDIM + c0];
#pragma unroll
    for (int j = 0; j < 32; ++j)
      sQ[row][c0 + j] = (f16)((tv[j] * inv + qeh[j]) * scale);
  }
  __syncthreads();

  // hoisted qn A-fragments (K blocks 0 and 32) — constant across key blocks
  V16 aq0, aq1;
  aq0.h[0] = *(const h8*)&sQ[16 * w + l16][16 * hv];
  aq0.h[1] = *(const h8*)&sQ[16 * w + l16][16 * hv + 8];
  aq1.h[0] = *(const h8*)&sQ[16 * w + l16][32 + 16 * hv];
  aq1.h[1] = *(const h8*)&sQ[16 * w + l16][40 + 16 * hv];

  const v8f vz = {0.f, 0.f, 0.f, 0.f, 0.f, 0.f, 0.f, 0.f};
  v8f o[4];
#pragma unroll
  for (int t = 0; t < 4; ++t) o[t] = vz;
  float mi[8], li[8];
#pragma unroll
  for (int r = 0; r < 8; ++r) { mi[r] = -1e30f; li[r] = 0.f; }

  const f16* bptr = &bias16[(size_t)h * NSEQ * NSEQ];

  for (int cb = 0; cb < NSEQ / 64; ++cb) {
    // ---- stage normalized K and transposed V, 4 threads per row ----
    {
      int row = tid >> 2, c0 = (tid & 3) * 16;
      const f16* ksrc = &qkv[(tok0 + cb * 64 + row) * (3 * DIMC) + (NHEAD + h) * HDIM + c0];
      float kv[16]; float ss = 0.f;
#pragma unroll
      for (int j = 0; j < 16; ++j) { float v = (float)ksrc[j]; kv[j] = v; ss += v * v; }
      ss += __shfl_xor(ss, 1);
      ss += __shfl_xor(ss, 2);
      float inv = 1.f / fmaxf(sqrtf(ss), 1e-12f);
#pragma unroll
      for (int j = 0; j < 16; ++j) sK[row][c0 + j] = (f16)(kv[j] * inv);
      const f16* vsrc = &qkv[(tok0 + cb * 64 + row) * (3 * DIMC) + (2 * NHEAD + h) * HDIM + c0];
#pragma unroll
      for (int j = 0; j < 16; ++j) sVT[c0 + j][row] = vsrc[j];
    }
    __syncthreads();

    // ---- S = qn @ kn^T : 4 tiles x 2 WMMAs (K=64) ----
    v8f s[4];
#pragma unroll
    for (int nb = 0; nb < 4; ++nb) {
      v16h b0 = *(const v16h*)&sK[nb * 16 + l16][16 * hv];
      v16h b1 = *(const v16h*)&sK[nb * 16 + l16][32 + 16 * hv];
      v8f a = vz;
      a = __builtin_amdgcn_wmma_f32_16x16x32_f16(false, aq0.v, false, b0, (short)0, a, false, false);
      a = __builtin_amdgcn_wmma_f32_16x16x32_f16(false, aq1.v, false, b1, (short)0, a, false, false);
      s[nb] = a;
    }

    // ---- add CPB relative-position bias ----
#pragma unroll
    for (int nb = 0; nb < 4; ++nb) {
      int col = cb * 64 + nb * 16 + l16;
#pragma unroll
      for (int r = 0; r < 8; ++r) {
        int rowg = qrow0 + 16 * w + r + 8 * hv;
        s[nb][r] += (float)bptr[(size_t)rowg * NSEQ + col];
      }
    }

    // ---- online softmax (rows live in 16-lane half-groups) ----
#pragma unroll
    for (int r = 0; r < 8; ++r) {
      float mx = fmaxf(fmaxf(s[0][r], s[1][r]), fmaxf(s[2][r], s[3][r]));
#pragma unroll
      for (int d = 8; d >= 1; d >>= 1) mx = fmaxf(mx, __shfl_xor(mx, d));
      float mnew = fmaxf(mi[r], mx);
      float alpha = __expf(mi[r] - mnew);
      float rs = 0.f;
#pragma unroll
      for (int nb = 0; nb < 4; ++nb) {
        float p = __expf(s[nb][r] - mnew);
        s[nb][r] = p;
        rs += p;
      }
#pragma unroll
      for (int d = 8; d >= 1; d >>= 1) rs += __shfl_xor(rs, d);
      li[r] = li[r] * alpha + rs;
      mi[r] = mnew;
#pragma unroll
      for (int t = 0; t < 4; ++t) o[t][r] *= alpha;
    }

    // ---- P roundtrip through LDS: D-layout -> A-layout (wave-private) ----
#pragma unroll
    for (int nb = 0; nb < 4; ++nb)
#pragma unroll
      for (int r = 0; r < 8; ++r)
        sP[w][r + 8 * hv][nb * 16 + l16] = (f16)s[nb][r];
    asm volatile("s_wait_dscnt 0" ::: "memory");   // RAW on wave-private LDS tile
    V16 p0, p1;
    p0.h[0] = *(const h8*)&sP[w][l16][16 * hv];
    p0.h[1] = *(const h8*)&sP[w][l16][16 * hv + 8];
    p1.h[0] = *(const h8*)&sP[w][l16][32 + 16 * hv];
    p1.h[1] = *(const h8*)&sP[w][l16][40 + 16 * hv];

    // ---- O += P @ V : 4 dim tiles x 2 WMMAs ----
#pragma unroll
    for (int dt = 0; dt < 4; ++dt) {
      v16h bv0 = *(const v16h*)&sVT[dt * 16 + l16][16 * hv];
      v16h bv1 = *(const v16h*)&sVT[dt * 16 + l16][32 + 16 * hv];
      o[dt] = __builtin_amdgcn_wmma_f32_16x16x32_f16(false, p0.v, false, bv0, (short)0, o[dt], false, false);
      o[dt] = __builtin_amdgcn_wmma_f32_16x16x32_f16(false, p1.v, false, bv1, (short)0, o[dt], false, false);
    }
    __syncthreads();
  }

  // ---- epilogue: divide by l, apply head gate, store f16 [tok][NH*HD] ----
#pragma unroll
  for (int r = 0; r < 8; ++r) {
    int rowg = qrow0 + 16 * w + r + 8 * hv;
    float g = masked[((size_t)b * NSEQ + rowg) * NHEAD + h] / li[r];
#pragma unroll
    for (int dt = 0; dt < 4; ++dt)
      attn_out[(tok0 + rowg) * DIMC + h * HDIM + dt * 16 + l16] = (f16)(o[dt][r] * g);
  }
}

// ------------------------------- launcher ----------------------------------

extern "C" void kernel_launch(void* const* d_in, const int* in_sizes, int n_in,
                              void* d_out, int out_size, void* d_ws, size_t ws_size,
                              hipStream_t stream) {
  const float* x      = (const float*)d_in[0];
  const float* rct    = (const float*)d_in[1];
  const float* qkv_w  = (const float*)d_in[2];
  const float* qkv_b  = (const float*)d_in[3];
  const float* qe     = (const float*)d_in[4];
  const float* temp   = (const float*)d_in[5];
  const float* proj_w = (const float*)d_in[6];
  const float* proj_b = (const float*)d_in[7];
  const float* cpb1_w = (const float*)d_in[8];
  const float* cpb1_b = (const float*)d_in[9];
  const float* cpb2_w = (const float*)d_in[10];
  const float* cpb2_b = (const float*)d_in[11];
  const float* wg_w   = (const float*)d_in[12];
  const float* wg0_w  = (const float*)d_in[13];
  const float* wg1_w  = (const float*)d_in[14];
  const int*   rpi    = (const int*)d_in[15];
  const int*   Hp     = (const int*)d_in[16];
  const int*   Wp     = (const int*)d_in[17];
  float* out = (float*)d_out;
  (void)in_sizes; (void)n_in; (void)out_size; (void)ws_size;

  char* ws = (char*)d_ws;
  size_t off = 0;
  auto alloc = [&](size_t bytes) -> void* {
    off = (off + 255) & ~(size_t)255;
    void* p = ws + off;
    off += bytes;
    return p;
  };
  f16*   x_h    = (f16*)alloc((size_t)TOK * DIMC * 2);             // 16 MB
  f16*   qw_h   = (f16*)alloc((size_t)3 * DIMC * DIMC * 2);        // 6 MB
  f16*   pw_h   = (f16*)alloc((size_t)DIMC * DIMC * 2);            // 2 MB
  f16*   qkv_h  = (f16*)alloc((size_t)TOK * 3 * DIMC * 2);         // 48 MB
  float* tbl    = (float*)alloc((size_t)TTBL * NHEAD * 4);         // 0.25 MB
  f16*   bias16 = (f16*)alloc((size_t)NHEAD * NSEQ * NSEQ * 2);    // 32 MB
  float* mk     = (float*)alloc((size_t)TOK * NHEAD * 4);          // 0.5 MB
  f16*   ao_h   = (f16*)alloc((size_t)TOK * DIMC * 2);             // 16 MB

  // 1) f32 -> f16 conversions for WMMA operands
  {
    int n = TOK * DIMC;
    k_f32_to_f16<<<(n + 255) / 256, 256, 0, stream>>>(x, x_h, n);
    n = 3 * DIMC * DIMC;
    k_f32_to_f16<<<(n + 255) / 256, 256, 0, stream>>>(qkv_w, qw_h, n);
    n = DIMC * DIMC;
    k_f32_to_f16<<<(n + 255) / 256, 256, 0, stream>>>(proj_w, pw_h, n);
  }

  // 2) CPB table + bias expansion (shared across batch)
  k_cpb<<<(TTBL * 32 + 255) / 256, 256, 0, stream>>>(rct, cpb1_w, cpb1_b, cpb2_w, cpb2_b, tbl);
  k_bias_expand<<<(NSEQ * NSEQ + 255) / 256, 256, 0, stream>>>(tbl, rpi, bias16);

  // 3) head gating
  k_gates<<<(TOK * 32 + 255) / 256, 256, 0, stream>>>(x, wg_w, wg0_w, wg1_w, mk);

  // 4) QKV projection: [8192,1024] x [3072,1024]^T -> f16 [8192,3072]
  {
    dim3 grid(TOK / 256, (3 * DIMC) / 64);
    k_gemm_f16<<<grid, 256, 0, stream>>>(x_h, qw_h, qkv_b, qkv_h, nullptr,
                                         TOK, 3 * DIMC, DIMC, 1);
  }

  // 5) cosine-sim flash attention with CPB bias + head gating
  {
    dim3 grid(NSEQ / 128, NHEAD, BATCH);
    k_attn<<<grid, 256, 0, stream>>>(qkv_h, bias16, qe, temp, mk, Hp, Wp, ao_h);
  }

  // 6) output projection: [8192,1024] x [1024,1024]^T + bias -> f32 out
  {
    dim3 grid(TOK / 256, DIMC / 64);
    k_gemm_f16<<<grid, 256, 0, stream>>>(ao_h, pw_h, proj_b, nullptr, out,
                                         TOK, DIMC, DIMC, 0);
  }
}
